// VisionTransformer_50268297233002
// MI455X (gfx1250) — matrix operand order repaired and checked by reading, not compiled, not run on hardware
//
#include <hip/hip_runtime.h>
#include <hip/hip_bf16.h>
#include <math.h>

// ---------------------------------------------------------------------------
// ViT-Base forward for MI455X (gfx1250).
// - All GEMMs on v_wmma_f32_16x16x32_bf16 (f32 accumulate, bf16 operands).
// - GEMM B-tiles staged into LDS via the Tensor Data Mover (tensor_load_to_lds,
//   double buffered, TENSORcnt-synchronized), with TDM LDS-padding used to
//   avoid bank conflicts on the ds_load_b128 fragment reads.
// ---------------------------------------------------------------------------

typedef __attribute__((ext_vector_type(16))) __bf16 bf16x16;
typedef __attribute__((ext_vector_type(8)))  float  f32x8;

union FragU { bf16x16 v; uint4 q[2]; };

__device__ __forceinline__ unsigned short f2bf(float f) {
  unsigned u = __float_as_uint(f);
  u += 0x7FFFu + ((u >> 16) & 1u);          // round-to-nearest-even
  return (unsigned short)(u >> 16);
}

#define WMMA_BF16(A, Bv, C) \
  __builtin_amdgcn_wmma_f32_16x16x32_bf16(false, (A), false, (Bv), (short)0, (C), false, false)

// ---- Tensor Data Mover availability (device pass only; host pass -> fallback)
#if defined(__has_builtin)
#if __has_builtin(__builtin_amdgcn_tensor_load_to_lds) && __has_builtin(__builtin_amdgcn_s_wait_tensorcnt)
#define USE_TDM 1
#endif
#endif
#ifndef USE_TDM
#define USE_TDM 0
#endif

#if USE_TDM
typedef __attribute__((ext_vector_type(4))) unsigned int v4u;
typedef __attribute__((ext_vector_type(4))) int v4i;
typedef __attribute__((ext_vector_type(8))) int v8i;

// Issue a TDM load of a 64-row x 128-elem bf16 tile (row pitch ldbt elements)
// into LDS at byte offset ldsOff, padding +4 DWORDs every 64 DWORDs so the
// LDS row pitch becomes 136 elements (bank-conflict-free b128 reads).
__device__ __forceinline__ void tdm_load_btile(const unsigned short* gsrc,
                                               unsigned ldsOff, int ldbt) {
  unsigned long long ga = (unsigned long long)(size_t)gsrc;
  // D# group0: count=1 | lds_addr | global_addr[56:0] | type=2
  v4u g0 = { 1u, ldsOff, (unsigned)ga,
             (unsigned)((ga >> 32) & 0x1FFFFFFu) | (2u << 30) };
  // D# group1: data_size=2B, pad_enable, pad_interval=64dw, pad_amount=4dw,
  // tensor_dim0=128, tensor_dim1=64, tile_dim0=128, tile_dim1=64, stride0=ldbt
  v8i g1 = { (int)((1u << 16) | (1u << 20) | (5u << 22) | (3u << 25)),
             (int)(128u << 16),          // tensor_dim0 low16 -> bits[63:48]
             (int)(64u << 16),           // tensor_dim1 low16 -> bits[95:80]
             (int)(128u << 16),          // tile_dim0 -> bits[127:112]
             64,                         // tile_dim1 -> bits[143:128]
             ldbt,                       // tensor_dim0_stride low32
             0, 0 };
  v4i zz = { 0, 0, 0, 0 };
#if defined(__clang_major__) && (__clang_major__ >= 23)
  v8i zz8 = { 0, 0, 0, 0, 0, 0, 0, 0 };
  __builtin_amdgcn_tensor_load_to_lds(g0, g1, zz, zz, zz8, 0);
#else
  __builtin_amdgcn_tensor_load_to_lds(g0, g1, zz, zz, 0);
#endif
}
#endif

// ---------------------------------------------------------------------------
// Patchify: images (32,3,224,224) f32 -> pbuf (6272 x 768) bf16
// ---------------------------------------------------------------------------
__global__ void k_patchify(const float* __restrict__ img, unsigned short* __restrict__ pbuf) {
  int i = blockIdx.x * 256 + threadIdx.x;
  if (i >= 6272 * 768) return;
  int k = i % 768, row = i / 768;
  int b = row / 196, p = row % 196;
  int c = k >> 8, rem = k & 255;
  int iy = rem >> 4, ix = rem & 15;
  int py = p / 14, px = p % 14;
  pbuf[i] = f2bf(img[(((size_t)b * 3 + c) * 224 + (py * 16 + iy)) * 224 + (px * 16 + ix)]);
}

// Transpose+convert f32 (R x C) -> bf16 (dstRows x dstCols), dst[n][k]=src[k][n]
__global__ void k_transpose(const float* __restrict__ src, unsigned short* __restrict__ dst,
                            int R, int C, int dstRows, int dstCols) {
  int i = blockIdx.x * 256 + threadIdx.x;
  if (i >= dstRows * dstCols) return;
  int r = i / dstCols;   // source column
  int c = i % dstCols;   // source row
  dst[i] = (r < C && c < R) ? f2bf(src[(size_t)c * C + r]) : (unsigned short)0;
}

// Per-head 64x64 transpose for Wq/Wk/Wv, all 12 layers at once.
__global__ void k_qkvw(const float* __restrict__ Wq, const float* __restrict__ Wk,
                       const float* __restrict__ Wv,
                       unsigned short* __restrict__ qT, unsigned short* __restrict__ kT,
                       unsigned short* __restrict__ vT) {
  int i = blockIdx.x * 256 + threadIdx.x;
  if (i >= 144 * 64 * 64) return;
  int d = i % 64, e = (i / 64) % 64, lh = i / 4096;
  size_t s = (size_t)lh * 4096 + (size_t)d * 64 + e;
  size_t o = (size_t)lh * 4096 + (size_t)e * 64 + d;
  qT[o] = f2bf(Wq[s]); kT[o] = f2bf(Wk[s]); vT[o] = f2bf(Wv[s]);
}

// cls row: x[b][0][n] = v_class[n] + pos(0,n)
__global__ void k_cls(const float* __restrict__ v_class, float* __restrict__ x) {
  int i = blockIdx.x * 256 + threadIdx.x;
  if (i >= 32 * 768) return;
  int b = i / 768, n = i % 768;
  x[(size_t)b * 197 * 768 + n] = v_class[n] + ((n & 1) ? 1.f : 0.f);
}

__global__ void k_getcls(const float* __restrict__ x, unsigned short* __restrict__ cls) {
  int i = blockIdx.x * 256 + threadIdx.x;
  if (i >= 32 * 768) return;
  int b = i / 768, n = i % 768;
  cls[i] = f2bf(x[(size_t)b * 197 * 768 + n]);
}

// ---------------------------------------------------------------------------
// WMMA GEMM with TDM-staged B tiles.
// C(MxN) = A(MxK,bf16,row-major) * B, with B given transposed (Bt: N x K).
// Block: 256 threads = 8 waves; wave w owns rows [blockM+16w,+16), N strip 64.
// K must be a multiple of 128. modes: 0 embed, 1 bias+GELU->bf16,
// 2 bias+residual->x, 3 bias->logits (n<1000)
// ---------------------------------------------------------------------------
__global__ __launch_bounds__(256) void k_gemm(
    const unsigned short* __restrict__ A, int lda,
    const unsigned short* __restrict__ Bt, int ldbt,
    const float* __restrict__ bias,
    float* __restrict__ outF, unsigned short* __restrict__ outB, int ldo,
    int M, int K, int mode) {
  // 64 N-rows x 128 K, row pitch 136 (TDM pad) -> conflict-free ds_load_b128
  __shared__ unsigned short Bs[2][64][136];

  const int tid = threadIdx.x;
  const int lane = tid & 31;
  const int wave = tid >> 5;
  const int l16 = lane & 15, lh = lane >> 4;
  const int nBase = blockIdx.x * 64;
  const int mBase = blockIdx.y * 128 + wave * 16;

  int arow = mBase + l16; if (arow > M - 1) arow = M - 1;
  const unsigned short* Ap = A + (size_t)arow * lda;

  f32x8 z = {0.f, 0.f, 0.f, 0.f, 0.f, 0.f, 0.f, 0.f};
  f32x8 acc[4] = {z, z, z, z};

  const int nChunks = K >> 7;
  const unsigned short* Bbase = Bt + (size_t)nBase * ldbt;

  // ---- stage chunk 0 ----
#if USE_TDM
  if (wave == 0) {
    tdm_load_btile(Bbase, (unsigned)(size_t)&Bs[0][0][0], ldbt);
    __builtin_amdgcn_s_wait_tensorcnt(0);
  }
#else
  {
#pragma unroll
    for (int v = 0; v < 4; ++v) {
      int e = tid * 32 + v * 8;
      int rr = e >> 7, cc = e & 127;
      *(uint4*)&Bs[0][rr][cc] = *(const uint4*)(Bbase + (size_t)rr * ldbt + cc);
    }
  }
#endif
  __syncthreads();

  for (int c = 0; c < nChunks; ++c) {
    const int st = c & 1;
    // stage next chunk into the other buffer (overlaps with compute below)
    if (c + 1 < nChunks) {
#if USE_TDM
      if (wave == 0)
        tdm_load_btile(Bbase + ((size_t)(c + 1) << 7),
                       (unsigned)(size_t)&Bs[st ^ 1][0][0], ldbt);
#else
#pragma unroll
      for (int v = 0; v < 4; ++v) {
        int e = tid * 32 + v * 8;
        int rr = e >> 7, cc = e & 127;
        *(uint4*)&Bs[st ^ 1][rr][cc] =
            *(const uint4*)(Bbase + ((size_t)(c + 1) << 7) + (size_t)rr * ldbt + cc);
      }
#endif
      __builtin_prefetch(Ap + ((c + 1) << 7), 0, 3);
    }

    const unsigned short* Ac = Ap + ((size_t)c << 7);
#pragma unroll
    for (int kk = 0; kk < 128; kk += 32) {
      FragU a, b0, b1, b2, b3;
      a.q[0] = *(const uint4*)(Ac + kk + lh * 16);
      a.q[1] = *(const uint4*)(Ac + kk + lh * 16 + 8);
      const unsigned short* B0 = &Bs[st][l16][kk + lh * 16];
      const unsigned short* B1 = &Bs[st][16 + l16][kk + lh * 16];
      const unsigned short* B2 = &Bs[st][32 + l16][kk + lh * 16];
      const unsigned short* B3 = &Bs[st][48 + l16][kk + lh * 16];
      b0.q[0] = *(const uint4*)(B0); b0.q[1] = *(const uint4*)(B0 + 8);
      b1.q[0] = *(const uint4*)(B1); b1.q[1] = *(const uint4*)(B1 + 8);
      b2.q[0] = *(const uint4*)(B2); b2.q[1] = *(const uint4*)(B2 + 8);
      b3.q[0] = *(const uint4*)(B3); b3.q[1] = *(const uint4*)(B3 + 8);
      acc[0] = WMMA_BF16(a.v, b0.v, acc[0]);
      acc[1] = WMMA_BF16(a.v, b1.v, acc[1]);
      acc[2] = WMMA_BF16(a.v, b2.v, acc[2]);
      acc[3] = WMMA_BF16(a.v, b3.v, acc[3]);
    }
    __syncthreads();            // everyone done reading Bs[st]
#if USE_TDM
    if (wave == 0 && c + 1 < nChunks) __builtin_amdgcn_s_wait_tensorcnt(0);
    __syncthreads();            // publish Bs[st^1]
#endif
  }

#pragma unroll
  for (int nt = 0; nt < 4; ++nt) {
    int n = nBase + nt * 16 + l16;
#pragma unroll
    for (int r = 0; r < 8; ++r) {
      int m = mBase + r + lh * 8;
      if (m >= M) continue;
      float v = acc[nt][r];
      if (mode == 0) {            // patch embed -> x rows 1..196 (+pos emb)
        int bb = m / 196, p = m - bb * 196, t = p + 1;
        float je = (float)(n & ~1);
        float ang = (float)t * __powf(10000.f, -je * (1.f / 197.f));
        float pos = (n & 1) ? __cosf(ang) : __sinf(ang);
        outF[((size_t)(bb * 197 + t)) * 768 + n] = v + bias[n] + pos;
      } else if (mode == 1) {     // bias + exact GELU -> bf16
        float g = v + bias[n];
        g = 0.5f * g * (1.f + erff(g * 0.70710678f));
        outB[(size_t)m * ldo + n] = f2bf(g);
      } else if (mode == 2) {     // bias + residual add into x
        float* p = outF + (size_t)m * ldo + n;
        *p = *p + v + bias[n];
      } else {                    // logits
        if (n < 1000) outF[(size_t)m * ldo + n] = v + bias[n];
      }
    }
  }
}

// ---------------------------------------------------------------------------
// QKV projection: per (mtile, b*h, sel) block of 128 threads (4 waves = 4 N-tiles)
// ---------------------------------------------------------------------------
__global__ __launch_bounds__(128) void k_qkv(
    const unsigned short* __restrict__ Hb,
    const unsigned short* __restrict__ WqT, const unsigned short* __restrict__ WkT,
    const unsigned short* __restrict__ WvT,
    const float* __restrict__ bq, const float* __restrict__ bk, const float* __restrict__ bv,
    unsigned short* __restrict__ Qb, unsigned short* __restrict__ Kb,
    unsigned short* __restrict__ Vb, int layer) {
  const int mt = blockIdx.x;             // 0..12
  const int bh = blockIdx.y;             // 0..383
  const int sel = blockIdx.z;            // 0..2
  const int bb = bh / 12, h = bh % 12;
  const int lane = threadIdx.x & 31, nt = threadIdx.x >> 5;
  const int l16 = lane & 15, lh = lane >> 4;

  const unsigned short* Wt = (sel == 0 ? WqT : sel == 1 ? WkT : WvT) +
                             (size_t)(layer * 12 + h) * 4096;
  const float* bias = (sel == 0 ? bq : sel == 1 ? bk : bv) + (size_t)(layer * 12 + h) * 64;
  unsigned short* Ob = (sel == 0 ? Qb : sel == 1 ? Kb : Vb) + (size_t)bh * 208 * 64;

  int t = mt * 16 + l16; int tc = t > 196 ? 196 : t;
  const unsigned short* Ap = Hb + ((size_t)(bb * 197 + tc)) * 768 + h * 64;

  f32x8 acc = {0.f, 0.f, 0.f, 0.f, 0.f, 0.f, 0.f, 0.f};
#pragma unroll
  for (int k0 = 0; k0 < 64; k0 += 32) {
    FragU a, w;
    a.q[0] = *(const uint4*)(Ap + k0 + lh * 16);
    a.q[1] = *(const uint4*)(Ap + k0 + lh * 16 + 8);
    const unsigned short* Bp = Wt + (size_t)(nt * 16 + l16) * 64 + k0 + lh * 16;
    w.q[0] = *(const uint4*)(Bp);
    w.q[1] = *(const uint4*)(Bp + 8);
    acc = WMMA_BF16(a.v, w.v, acc);
  }
#pragma unroll
  for (int r = 0; r < 8; ++r) {
    int tr = mt * 16 + r + lh * 8;
    int e = nt * 16 + l16;
    Ob[(size_t)tr * 64 + e] = f2bf(acc[r] + bias[e]);
  }
}

// ---------------------------------------------------------------------------
// Fused attention: one wave per (b,h, 16-query tile).
// ---------------------------------------------------------------------------
__global__ __launch_bounds__(32) void k_attn(
    const unsigned short* __restrict__ Qb, const unsigned short* __restrict__ Kb,
    const unsigned short* __restrict__ Vb, float* __restrict__ x) {
  __shared__ float          sc[16][224];
  __shared__ unsigned short ab[16][224];
  __shared__ unsigned short vt[64][224];

  const int idx = blockIdx.x;
  const int qt = idx % 13, bh = idx / 13;
  const int bb = bh / 12, h = bh % 12;
  const int lane = threadIdx.x;
  const int l16 = lane & 15, lh = lane >> 4;

  const unsigned short* Q  = Qb + (size_t)bh * 208 * 64;
  const unsigned short* Kp = Kb + (size_t)bh * 208 * 64;
  const unsigned short* Vp = Vb + (size_t)bh * 208 * 64;

  for (int i = lane; i < 64 * 224; i += 32) {
    int d = i / 224, kt = i % 224;
    vt[d][kt] = (kt < 197) ? Vp[(size_t)kt * 64 + d] : (unsigned short)0;
  }

  FragU qa[2];
  const unsigned short* Qr = Q + (size_t)(qt * 16 + l16) * 64;
#pragma unroll
  for (int s = 0; s < 2; ++s) {
    qa[s].q[0] = *(const uint4*)(Qr + s * 32 + lh * 16);
    qa[s].q[1] = *(const uint4*)(Qr + s * 32 + lh * 16 + 8);
  }

  for (int j = 0; j < 13; ++j) {
    f32x8 acc = {0.f, 0.f, 0.f, 0.f, 0.f, 0.f, 0.f, 0.f};
    const unsigned short* Kr = Kp + (size_t)(j * 16 + l16) * 64;
#pragma unroll
    for (int s = 0; s < 2; ++s) {
      FragU kb;
      kb.q[0] = *(const uint4*)(Kr + s * 32 + lh * 16);
      kb.q[1] = *(const uint4*)(Kr + s * 32 + lh * 16 + 8);
      acc = WMMA_BF16(qa[s].v, kb.v, acc);
    }
#pragma unroll
    for (int r = 0; r < 8; ++r) sc[r + lh * 8][j * 16 + l16] = acc[r] * 0.125f;
  }
  __syncthreads();

  if (lane < 16) {
    float mx = -3.4e38f;
    for (int c = 0; c < 197; ++c) mx = fmaxf(mx, sc[lane][c]);
    float sum = 0.f;
    for (int c = 0; c < 197; ++c) { float e = __expf(sc[lane][c] - mx); sc[lane][c] = e; sum += e; }
    float inv = 1.f / sum;
    for (int c = 0; c < 224; ++c)
      ab[lane][c] = (c < 197) ? f2bf(sc[lane][c] * inv) : (unsigned short)0;
  }
  __syncthreads();

  f32x8 z = {0.f, 0.f, 0.f, 0.f, 0.f, 0.f, 0.f, 0.f};
  f32x8 o[4] = {z, z, z, z};
#pragma unroll
  for (int s = 0; s < 7; ++s) {
    FragU pa;
    const unsigned short* Ar = &ab[l16][0];
    pa.q[0] = *(const uint4*)(Ar + s * 32 + lh * 16);
    pa.q[1] = *(const uint4*)(Ar + s * 32 + lh * 16 + 8);
#pragma unroll
    for (int nt = 0; nt < 4; ++nt) {
      FragU vb2;
      const unsigned short* Br = &vt[nt * 16 + l16][0] + s * 32 + lh * 16;
      vb2.q[0] = *(const uint4*)(Br);
      vb2.q[1] = *(const uint4*)(Br + 8);
      o[nt] = WMMA_BF16(pa.v, vb2.v, o[nt]);
    }
  }
#pragma unroll
  for (int nt = 0; nt < 4; ++nt) {
#pragma unroll
    for (int r = 0; r < 8; ++r) {
      int t = qt * 16 + r + lh * 8;
      if (t < 197) {
        float* p = x + ((size_t)(bb * 197 + t)) * 768 + h * 64 + nt * 16 + l16;
        *p += o[nt][r];
      }
    }
  }
}

// ---------------------------------------------------------------------------
// LayerNorm over 768, one block (256 thr) per token row; bf16 output.
// ---------------------------------------------------------------------------
__global__ __launch_bounds__(256) void k_ln(const float* __restrict__ x,
                                            const float* __restrict__ gamma,
                                            const float* __restrict__ beta,
                                            unsigned short* __restrict__ dst) {
  __shared__ float r1[256], r2[256];
  const int row = blockIdx.x;
  const float* xr = x + (size_t)row * 768;
  float s = 0.f, s2 = 0.f;
  for (int c = threadIdx.x; c < 768; c += 256) { float v = xr[c]; s += v; s2 += v * v; }
  r1[threadIdx.x] = s; r2[threadIdx.x] = s2;
  __syncthreads();
  for (int o = 128; o > 0; o >>= 1) {
    if ((int)threadIdx.x < o) { r1[threadIdx.x] += r1[threadIdx.x + o]; r2[threadIdx.x] += r2[threadIdx.x + o]; }
    __syncthreads();
  }
  float mean = r1[0] * (1.f / 768.f);
  float var  = r2[0] * (1.f / 768.f) - mean * mean;
  float rs = rsqrtf(var + 1e-5f);
  for (int c = threadIdx.x; c < 768; c += 256)
    dst[(size_t)row * 768 + c] = f2bf((xr[c] - mean) * rs * gamma[c] + beta[c]);
}

// Final row softmax over 1000 classes, one block per batch.
__global__ __launch_bounds__(256) void k_softmax(const float* __restrict__ logits,
                                                 float* __restrict__ out) {
  __shared__ float red[256];
  const int b = blockIdx.x;
  const float* lr = logits + (size_t)b * 1008;
  float mx = -3.4e38f;
  for (int c = threadIdx.x; c < 1000; c += 256) mx = fmaxf(mx, lr[c]);
  red[threadIdx.x] = mx; __syncthreads();
  for (int o = 128; o > 0; o >>= 1) {
    if ((int)threadIdx.x < o) red[threadIdx.x] = fmaxf(red[threadIdx.x], red[threadIdx.x + o]);
    __syncthreads();
  }
  mx = red[0]; __syncthreads();
  float s = 0.f;
  for (int c = threadIdx.x; c < 1000; c += 256) s += __expf(lr[c] - mx);
  red[threadIdx.x] = s; __syncthreads();
  for (int o = 128; o > 0; o >>= 1) {
    if ((int)threadIdx.x < o) red[threadIdx.x] += red[threadIdx.x + o];
    __syncthreads();
  }
  float inv = 1.f / red[0];
  for (int c = threadIdx.x; c < 1000; c += 256)
    out[(size_t)b * 1000 + c] = __expf(lr[c] - mx) * inv;
}

// ---------------------------------------------------------------------------
extern "C" void kernel_launch(void* const* d_in, const int* in_sizes, int n_in,
                              void* d_out, int out_size, void* d_ws, size_t ws_size,
                              hipStream_t stream) {
  (void)in_sizes; (void)n_in; (void)out_size; (void)ws_size;
  const float* images  = (const float*)d_in[0];
  const float* W_embed = (const float*)d_in[1];
  const float* b_embed = (const float*)d_in[2];
  const float* v_class = (const float*)d_in[3];
  const float* ln1_g   = (const float*)d_in[4];
  const float* ln1_b   = (const float*)d_in[5];
  const float* ln2_g   = (const float*)d_in[6];
  const float* ln2_b   = (const float*)d_in[7];
  const float* Wq      = (const float*)d_in[8];
  const float* bq      = (const float*)d_in[9];
  const float* Wk      = (const float*)d_in[10];
  const float* bk      = (const float*)d_in[11];
  const float* Wv      = (const float*)d_in[12];
  const float* bv      = (const float*)d_in[13];
  const float* Wm1     = (const float*)d_in[14];
  const float* bm1     = (const float*)d_in[15];
  const float* Wm2     = (const float*)d_in[16];
  const float* bm2     = (const float*)d_in[17];
  const float* Wout    = (const float*)d_in[18];
  const float* bout    = (const float*)d_in[19];
  float* out = (float*)d_out;

  char* w = (char*)d_ws;
  size_t off = 0;
  auto carve = [&](size_t bytes) -> void* {
    void* p = w + off; off += (bytes + 255) & ~(size_t)255; return p;
  };
  unsigned short* pbuf  = (unsigned short*)carve((size_t)6272 * 768 * 2);
  unsigned short* wembT = (unsigned short*)carve((size_t)768 * 768 * 2);
  float*          x     = (float*)carve((size_t)32 * 197 * 768 * 4);
  unsigned short* hbuf  = (unsigned short*)carve((size_t)6304 * 768 * 2);
  unsigned short* wqT   = (unsigned short*)carve((size_t)144 * 4096 * 2);
  unsigned short* wkT   = (unsigned short*)carve((size_t)144 * 4096 * 2);
  unsigned short* wvT   = (unsigned short*)carve((size_t)144 * 4096 * 2);
  unsigned short* qbuf  = (unsigned short*)carve((size_t)384 * 208 * 64 * 2);
  unsigned short* kbuf  = (unsigned short*)carve((size_t)384 * 208 * 64 * 2);
  unsigned short* vbuf  = (unsigned short*)carve((size_t)384 * 208 * 64 * 2);
  unsigned short* wm1T  = (unsigned short*)carve((size_t)3072 * 768 * 2);
  unsigned short* wm2T  = (unsigned short*)carve((size_t)768 * 3072 * 2);
  unsigned short* m1buf = (unsigned short*)carve((size_t)6304 * 3072 * 2);
  unsigned short* woutT = (unsigned short*)carve((size_t)1024 * 768 * 2);
  unsigned short* clsb  = (unsigned short*)carve((size_t)32 * 768 * 2);
  float*          logit = (float*)carve((size_t)32 * 1008 * 4);

  // ---- setup / one-time conversions ----
  k_patchify<<<(6272 * 768 + 255) / 256, 256, 0, stream>>>(images, pbuf);
  k_transpose<<<(768 * 768 + 255) / 256, 256, 0, stream>>>(W_embed, wembT, 768, 768, 768, 768);
  k_qkvw<<<(144 * 4096 + 255) / 256, 256, 0, stream>>>(Wq, Wk, Wv, wqT, wkT, wvT);
  k_transpose<<<(1024 * 768 + 255) / 256, 256, 0, stream>>>(Wout, woutT, 768, 1000, 1024, 768);
  k_cls<<<(32 * 768 + 255) / 256, 256, 0, stream>>>(v_class, x);
  k_gemm<<<dim3(768 / 64, 6272 / 128), 256, 0, stream>>>(
      pbuf, 768, wembT, 768, b_embed, x, nullptr, 768, 6272, 768, 0);

  // ---- transformer blocks ----
  for (int l = 0; l < 12; ++l) {
    k_ln<<<6304, 256, 0, stream>>>(x, ln1_g + l * 768, ln1_b + l * 768, hbuf);
    k_qkv<<<dim3(13, 384, 3), 128, 0, stream>>>(hbuf, wqT, wkT, wvT, bq, bk, bv,
                                                qbuf, kbuf, vbuf, l);
    k_attn<<<384 * 13, 32, 0, stream>>>(qbuf, kbuf, vbuf, x);
    k_ln<<<6304, 256, 0, stream>>>(x, ln2_g + l * 768, ln2_b + l * 768, hbuf);
    k_transpose<<<(3072 * 768 + 255) / 256, 256, 0, stream>>>(
        Wm1 + (size_t)l * 768 * 3072, wm1T, 768, 3072, 3072, 768);
    k_transpose<<<(768 * 3072 + 255) / 256, 256, 0, stream>>>(
        Wm2 + (size_t)l * 3072 * 768, wm2T, 3072, 768, 768, 3072);
    k_gemm<<<dim3(3072 / 64, (6304 + 127) / 128), 256, 0, stream>>>(
        hbuf, 768, wm1T, 768, bm1 + (size_t)l * 3072, nullptr, m1buf, 3072, 6304, 768, 1);
    k_gemm<<<dim3(768 / 64, (6304 + 127) / 128), 256, 0, stream>>>(
        m1buf, 3072, wm2T, 3072, bm2 + (size_t)l * 768, x, nullptr, 768, 6304, 3072, 2);
  }

  // ---- classifier head ----
  k_getcls<<<(32 * 768 + 255) / 256, 256, 0, stream>>>(x, clsb);
  k_gemm<<<dim3(16, 1), 256, 0, stream>>>(
      clsb, 768, woutT, 768, bout, logit, nullptr, 1008, 32, 768, 3);
  k_softmax<<<32, 256, 0, stream>>>(logit, out);
}
